// RNN_55671366090804
// MI455X (gfx1250) — matrix operand order repaired and checked by reading, not compiled, not run on hardware
//
#include <hip/hip_runtime.h>

// ---------------------------------------------------------------------------
// Fused 2-layer LSTM (B=2048, T=512, IN=20, H=64) + final 64->5 projection.
// One block = 16 batch rows, 4 waves (wave32). Wave w owns hidden columns
// [16w, 16w+16) across all four gates. Weights live in registers as
// pre-swizzled WMMA B-fragments; biases live in persistent v8f accumulator-
// layout registers fed straight into the WMMA C operand. h0/h1 are exchanged
// via double-buffered LDS. Matmuls: v_wmma_f32_16x16x32_f16 (f16 in, f32
// accumulate). Cell state and nonlinearities are f32; activations use native
// v_tanh_f32 when available, else v_exp_f32 + v_rcp_f32.
// ---------------------------------------------------------------------------

typedef __attribute__((ext_vector_type(16))) _Float16 v16h;
typedef __attribute__((ext_vector_type(8)))  _Float16 v8h;
typedef __attribute__((ext_vector_type(8)))  float    v8f;

#define HID   64
#define NG    4
#define INPUT 20          // compile-time input width (reference INPUT_SIZE)

#define WMMA_F16(a, b, c) \
  __builtin_amdgcn_wmma_f32_16x16x32_f16(false, (a), false, (b), (short)0, (c), false, false)

#if __has_builtin(__builtin_amdgcn_tanhf)
// CDNA5 has TANH in the TRANS32 unit: 1 op per tanh, 2 per sigmoid.
__device__ __forceinline__ float tanh_fast(float x) { return __builtin_amdgcn_tanhf(x); }
__device__ __forceinline__ float sigm(float x) {
  return __builtin_fmaf(0.5f, __builtin_amdgcn_tanhf(0.5f * x), 0.5f);
}
#else
// Fallback: v_exp_f32 + raw v_rcp_f32 (no IEEE div_scale/div_fixup sequence).
__device__ __forceinline__ float fast_rcp(float x)  { return __builtin_amdgcn_rcpf(x); }
__device__ __forceinline__ float sigm(float x)      { return fast_rcp(1.0f + __expf(-x)); }
__device__ __forceinline__ float tanh_fast(float x) { return 1.0f - 2.0f * fast_rcp(1.0f + __expf(2.0f * x)); }
#endif

// B-fragment (32x16, f16) for 16x16x32 WMMA.
// Lane L: n = L%16, hi = L/16; elems 0..7 hold K = kbase+8*hi+e, elems 8..15 hold K = kbase+16+8*hi+(e-8).
__device__ __forceinline__ v16h b_frag_load(const float* __restrict__ W, int kstride,
                                            int row_base, int kbase, int kvalid, int lane) {
  const int n  = lane & 15;
  const int kb = lane & 16;           // 0 or 16
  const float* row = W + (size_t)(row_base + n) * kstride;
  v16h b;
#pragma unroll
  for (int e = 0; e < 16; ++e) {
    int k = kbase + (kb >> 1) + (e & 7) + ((e & 8) << 1);
    b[e] = (k < kvalid) ? (_Float16)row[k] : (_Float16)0.0f;
  }
  return b;
}

// A-fragment (16x32, f16) from LDS h[16][64], K window [kbase, kbase+32).
// Lane L: m = L%16, hi = L/16; elems 0..7 -> K = kbase+8*hi+e,
// elems 8..15 -> K = kbase+16+8*hi+(e-8). Both halves are 16B-aligned.
__device__ __forceinline__ v16h a_frag_lds(const _Float16* __restrict__ h,
                                           int m, int hi, int kbase) {
  const _Float16* p = h + m * HID + kbase + 8 * hi;
  v8h lo = *(const v8h*)p;
  v8h hh = *(const v8h*)(p + 16);
  v16h a;
#pragma unroll
  for (int e = 0; e < 8; ++e) { a[e] = lo[e]; a[8 + e] = hh[e]; }
  return a;
}

// A-fragment for x_t (K = INPUT=20 padded to 32), from global, vectorized.
// elems 0..7:  k = 8*hi + e      -> always < 20: two unconditional float4.
// elems 8..11: k = 16 + 8*hi + e -> valid iff hi==0 (k=16..19): masked float4.
// elems 12..15: k >= 20 always   -> compile-time zero.
__device__ __forceinline__ v16h x_frag_load(const float* __restrict__ row, int hi) {
  const float4 a0 = *(const float4*)(row + 8 * hi);
  const float4 a1 = *(const float4*)(row + 8 * hi + 4);
  v16h a;
  a[0] = (_Float16)a0.x; a[1] = (_Float16)a0.y;
  a[2] = (_Float16)a0.z; a[3] = (_Float16)a0.w;
  a[4] = (_Float16)a1.x; a[5] = (_Float16)a1.y;
  a[6] = (_Float16)a1.z; a[7] = (_Float16)a1.w;
  if (hi == 0) {
    const float4 a2 = *(const float4*)(row + 16);
    a[8]  = (_Float16)a2.x; a[9]  = (_Float16)a2.y;
    a[10] = (_Float16)a2.z; a[11] = (_Float16)a2.w;
  } else {
    a[8] = (_Float16)0.0f; a[9] = (_Float16)0.0f;
    a[10] = (_Float16)0.0f; a[11] = (_Float16)0.0f;
  }
  a[12] = (_Float16)0.0f; a[13] = (_Float16)0.0f;
  a[14] = (_Float16)0.0f; a[15] = (_Float16)0.0f;
  return a;
}

__global__ __launch_bounds__(128, 1)
void lstm2_fused_wmma(const float* __restrict__ x,
                      const float* __restrict__ w_ih0, const float* __restrict__ w_hh0,
                      const float* __restrict__ b_ih0, const float* __restrict__ b_hh0,
                      const float* __restrict__ w_ih1, const float* __restrict__ w_hh1,
                      const float* __restrict__ b_ih1, const float* __restrict__ b_hh1,
                      const float* __restrict__ w_out, const float* __restrict__ b_out,
                      float* __restrict__ out, int B, int T) {
  __shared__ __align__(16) _Float16 h0buf[2][16][HID];  // layer-0 h (== layer-1 input y0)
  __shared__ __align__(16) _Float16 h1buf[2][16][HID];  // layer-1 h
  __shared__ __align__(16) float    hfin[16][HID];      // final h1 (f32) for projection

  const int tid   = threadIdx.x;
  const int lane  = tid & 31;
  const int wv    = tid >> 5;          // 0..3: hidden slice
  const int m     = lane & 15;         // batch row within tile (A-frag / C rows)
  const int hi    = (lane >> 4) & 1;
  const int nloc  = lane & 15;         // C-layout column within tile
  const int nglob = 16 * wv + nloc;    // global hidden column owned by this lane
  const int batch0 = blockIdx.x * 16;

  // Zero-init both h double-buffers (h=0, c=0 at t=0).
  for (int i = tid; i < 2 * 16 * HID; i += blockDim.x) {
    (&h0buf[0][0][0])[i] = (_Float16)0.0f;
    (&h1buf[0][0][0])[i] = (_Float16)0.0f;
  }

  // ---- One-time: weight B-fragments into registers (reused T times) -------
  v16h Bih0[NG];          // layer0 W_ih: K = 20 (padded to 32)
  v16h Bhh0[NG][2];       // layer0 W_hh: K = 64 -> 2 chunks
  v16h Bih1[NG][2];       // layer1 W_ih: K = 64
  v16h Bhh1[NG][2];       // layer1 W_hh: K = 64
  // Biases pre-splatted into the accumulator layout; fed as the C operand of
  // the first WMMA per gate -> no per-iteration broadcast movs.
  v8f biasv0[NG], biasv1[NG];
#pragma unroll
  for (int g = 0; g < NG; ++g) {
    const int rbase = g * HID + 16 * wv;       // gate-major row block
    Bih0[g] = b_frag_load(w_ih0, INPUT, rbase, 0, INPUT, lane);
#pragma unroll
    for (int c = 0; c < 2; ++c) {
      Bhh0[g][c] = b_frag_load(w_hh0, HID, rbase, 32 * c, HID, lane);
      Bih1[g][c] = b_frag_load(w_ih1, HID, rbase, 32 * c, HID, lane);
      Bhh1[g][c] = b_frag_load(w_hh1, HID, rbase, 32 * c, HID, lane);
    }
    const float b0 = b_ih0[rbase + nloc] + b_hh0[rbase + nloc];
    const float b1 = b_ih1[rbase + nloc] + b_hh1[rbase + nloc];
#pragma unroll
    for (int r = 0; r < 8; ++r) { biasv0[g][r] = b0; biasv1[g][r] = b1; }
  }

  // Cell states live in the WMMA C-layout across the whole sequence.
  v8f c0, c1;
#pragma unroll
  for (int r = 0; r < 8; ++r) { c0[r] = 0.0f; c1[r] = 0.0f; }
  float h1last[8];
#pragma unroll
  for (int r = 0; r < 8; ++r) h1last[r] = 0.0f;

  int brow = batch0 + m; if (brow >= B) brow = B - 1;
  const float* xrow = x + (size_t)brow * T * INPUT;

  __syncthreads();

  v16h xA = x_frag_load(xrow, hi);   // t = 0

  for (int t = 0; t < T; ++t) {
    const int rbuf = t & 1, wbuf = rbuf ^ 1;

    // Prefetch next timestep's x fragment (hides global latency under WMMA).
    const int tn = (t + 1 < T) ? (t + 1) : t;
    v16h xN = x_frag_load(xrow + (size_t)tn * INPUT, hi);

    // ---------------- Layer 0: gates = x@Wih0^T + h0@Whh0^T + b ------------
    v16h hA0a = a_frag_lds(&h0buf[rbuf][0][0], m, hi, 0);
    v16h hA0b = a_frag_lds(&h0buf[rbuf][0][0], m, hi, 32);

    v8f acc0[NG];
#pragma unroll
    for (int g = 0; g < NG; ++g) {
      v8f a = WMMA_F16(xA, Bih0[g], biasv0[g]);   // C = persistent bias regs
      a = WMMA_F16(hA0a, Bhh0[g][0], a);
      a = WMMA_F16(hA0b, Bhh0[g][1], a);
      acc0[g] = a;
    }
    // Per-lane LSTM cell update (i,f,g,o share (m,n) in this lane).
#pragma unroll
    for (int r = 0; r < 8; ++r) {
      float ig = sigm(acc0[0][r]);
      float fg = sigm(acc0[1][r]);
      float gg = tanh_fast(acc0[2][r]);
      float og = sigm(acc0[3][r]);
      float cc = fg * c0[r] + ig * gg;
      c0[r] = cc;
      h0buf[wbuf][r + 8 * hi][nglob] = (_Float16)(og * tanh_fast(cc));
    }
    __syncthreads();   // y0_t visible to all waves; protects h1buf[rbuf] reads

    // ---------------- Layer 1: gates = y0@Wih1^T + h1@Whh1^T + b -----------
    v16h yAa  = a_frag_lds(&h0buf[wbuf][0][0], m, hi, 0);
    v16h yAb  = a_frag_lds(&h0buf[wbuf][0][0], m, hi, 32);
    v16h hA1a = a_frag_lds(&h1buf[rbuf][0][0], m, hi, 0);
    v16h hA1b = a_frag_lds(&h1buf[rbuf][0][0], m, hi, 32);

    v8f acc1[NG];
#pragma unroll
    for (int g = 0; g < NG; ++g) {
      v8f a = WMMA_F16(yAa, Bih1[g][0], biasv1[g]);   // C = persistent bias regs
      a = WMMA_F16(yAb,  Bih1[g][1], a);
      a = WMMA_F16(hA1a, Bhh1[g][0], a);
      a = WMMA_F16(hA1b, Bhh1[g][1], a);
      acc1[g] = a;
    }
#pragma unroll
    for (int r = 0; r < 8; ++r) {
      float ig = sigm(acc1[0][r]);
      float fg = sigm(acc1[1][r]);
      float gg = tanh_fast(acc1[2][r]);
      float og = sigm(acc1[3][r]);
      float cc = fg * c1[r] + ig * gg;
      c1[r] = cc;
      float hh = og * tanh_fast(cc);
      h1last[r] = hh;
      h1buf[wbuf][r + 8 * hi][nglob] = (_Float16)hh;
    }
    xA = xN;
    __syncthreads();   // h1_t visible; protects next step's buffer reuse
  }

  // ---- Final projection: out[b, :] = h1_last @ w_out^T + b_out (f32) ------
#pragma unroll
  for (int r = 0; r < 8; ++r) hfin[r + 8 * hi][nglob] = h1last[r];
  __syncthreads();

  if (tid < 16 * 5) {
    const int mrow = tid / 5, oo = tid % 5;
    const int b = batch0 + mrow;
    if (b < B) {
      float s = b_out[oo];
      const float* wr = w_out + (size_t)oo * HID;
#pragma unroll 8
      for (int k = 0; k < HID; ++k) s += hfin[mrow][k] * wr[k];
      out[(size_t)b * 5 + oo] = s;
    }
  }
}

extern "C" void kernel_launch(void* const* d_in, const int* in_sizes, int n_in,
                              void* d_out, int out_size, void* d_ws, size_t ws_size,
                              hipStream_t stream) {
  const float* x     = (const float*)d_in[0];
  const float* w_ih0 = (const float*)d_in[1];
  const float* w_hh0 = (const float*)d_in[2];
  const float* b_ih0 = (const float*)d_in[3];
  const float* b_hh0 = (const float*)d_in[4];
  const float* w_ih1 = (const float*)d_in[5];
  const float* w_hh1 = (const float*)d_in[6];
  const float* b_ih1 = (const float*)d_in[7];
  const float* b_hh1 = (const float*)d_in[8];
  const float* w_out = (const float*)d_in[9];
  const float* b_out = (const float*)d_in[10];

  const int OUTD = 5;
  const int B = out_size / OUTD;                 // 2048
  const int T = in_sizes[0] / (B * INPUT);       // 512

  const int blocks = (B + 15) / 16;              // 128 blocks x 128 threads
  lstm2_fused_wmma<<<blocks, 128, 0, stream>>>(
      x, w_ih0, w_hh0, b_ih0, b_hh0,
      w_ih1, w_hh1, b_ih1, b_hh1,
      w_out, b_out, (float*)d_out, B, T);
}
